// MultiHeadedAttention_14310831030805
// MI455X (gfx1250) — matrix operand order repaired and checked by reading, not compiled, not run on hardware
//
#include <hip/hip_runtime.h>

// ---------------------------------------------------------------------------
// MultiHeadedAttention for MI455X (gfx1250): bf16 WMMA, flash-attention core,
// async global->LDS staging (CDNA5 ASYNCcnt path) for K/V tiles.
// B=4, D=1024, N=2048, H=16, HD=64.
// ---------------------------------------------------------------------------

typedef __attribute__((ext_vector_type(16))) __bf16 v16bf;
typedef __attribute__((ext_vector_type(8)))  float  v8f;

union Frag { v16bf v; unsigned u[8]; };

#define Bc 4
#define Dc 1024
#define Nc 2048
#define Hc 16
#define HDc 64

__device__ __forceinline__ unsigned short f2bf(float f) {
    unsigned u = __float_as_uint(f);
    unsigned rnd = 0x7FFFu + ((u >> 16) & 1u);
    return (unsigned short)((u + rnd) >> 16);
}

// Async bulk copy: global -> LDS, 16 bytes, tracked by ASYNCcnt.
// Low 32 bits of a generic pointer into __shared__ are the LDS byte address
// (ISA 10.2: LDS aperture truncates to addr[31:0]).
__device__ __forceinline__ void async_ld_b128(void* lds_ptr, const void* gptr) {
    unsigned loff = (unsigned)(size_t)lds_ptr;
    asm volatile("global_load_async_to_lds_b128 %0, %1, off"
                 :: "v"(loff), "v"(gptr) : "memory");
}
__device__ __forceinline__ void wait_async0() {
    asm volatile("s_wait_asynccnt 0x0" ::: "memory");
}

// A-fragment (16x32, 16-bit): lane row = lane&15, half = lane>>4.
// v[j]   holds K = 8*half + 2j {,+1};  v[4+j] holds K = 16 + 8*half + 2j {,+1}.
__device__ __forceinline__ v16bf load_frag_a(const unsigned short* a, int stride,
                                             int kbase, int lane) {
    const int row = lane & 15, half = lane >> 4;
    const unsigned short* base = a + row * stride + kbase + half * 8;
    Frag f;
#pragma unroll
    for (int j = 0; j < 4; ++j) {
        f.u[j]     = *(const unsigned*)(base + 2 * j);
        f.u[j + 4] = *(const unsigned*)(base + 16 + 2 * j);
    }
    return f.v;
}

// B-fragment (32x16, 16-bit) from a transposed tile bt[N][K] (row stride = stride):
// lane col = lane&15, half = lane>>4; v[j] holds K = 8*half + 2j {,+1};  v[4+j] holds K = 16 + 8*half + 2j {,+1}.
__device__ __forceinline__ v16bf load_frag_b_t(const unsigned short* bt, int stride,
                                               int kbase, int lane) {
    const int col = lane & 15, half = lane >> 4;
    const unsigned short* base = bt + col * stride + kbase + half * 8;
    Frag f;
#pragma unroll
    for (int j = 0; j < 4; ++j) {
        f.u[j]     = *(const unsigned*)(base + 2 * j);
        f.u[j + 4] = *(const unsigned*)(base + 16 + 2 * j);
    }
    return f.v;
}

// ---------------------------------------------------------------------------
// Kernel 1: QKV projection GEMM. out_e[n] = sum_d W[e,d]*x[b,d,n] + bias[e].
// mode 0: Q -> [b][h][n][hd] bf16, scaled by 1/sqrt(HD)=0.125
// mode 1: K -> [b][h][n][hd] bf16
// mode 2: V -> [b][h][hd][n] bf16
// Tile: 128 (e) x 64 (n) per WG, 8 waves each 32x32, K-step 32.
// ---------------------------------------------------------------------------
__global__ __launch_bounds__(256)
void qkv_proj_kernel(const float* __restrict__ X, const float* __restrict__ W,
                     const float* __restrict__ bias, unsigned short* __restrict__ out,
                     int mode) {
    __shared__ __align__(16) unsigned short ldsW[128 * 32];
    __shared__ __align__(16) unsigned short ldsX[64 * 32];

    const int b  = blockIdx.z;
    const int e0 = blockIdx.y * 128;
    const int n0 = blockIdx.x * 64;
    const int t = threadIdx.x, lane = t & 31, w = t >> 5;
    const int wr = w >> 1, wc = w & 1;            // 4x2 wave grid
    const int half = lane >> 4, col = lane & 15;

    v8f acc[2][2] = {};

    for (int k0 = 0; k0 < Dc; k0 += 32) {
        __syncthreads();
        // Prefetch next K-tile while this tile computes
        if (k0 + 32 < Dc) {
            __builtin_prefetch(&W[(size_t)(e0 + (t >> 1)) * Dc + k0 + 32], 0, 0);
            __builtin_prefetch(&X[(size_t)b * Dc * Nc + (size_t)(k0 + 32 + (t >> 3)) * Nc + n0],
                               0, 0);
        }
        // stage W tile [128][32] fp32 -> bf16
#pragma unroll
        for (int i = 0; i < 16; ++i) {
            int idx = t + i * 256;
            int r = idx >> 5, c = idx & 31;
            ldsW[idx] = f2bf(W[(size_t)(e0 + r) * Dc + k0 + c]);
        }
        // stage X tile transposed: ldsX[n][k]
#pragma unroll
        for (int i = 0; i < 8; ++i) {
            int idx = t + i * 256;
            int k = idx >> 6, n = idx & 63;
            ldsX[n * 32 + k] =
                f2bf(X[(size_t)b * Dc * Nc + (size_t)(k0 + k) * Nc + n0 + n]);
        }
        __syncthreads();

        v16bf a0 = load_frag_a(ldsW + (wr * 32) * 32, 32, 0, lane);
        v16bf a1 = load_frag_a(ldsW + (wr * 32 + 16) * 32, 32, 0, lane);
        v16bf b0 = load_frag_b_t(ldsX + (wc * 32) * 32, 32, 0, lane);
        v16bf b1 = load_frag_b_t(ldsX + (wc * 32 + 16) * 32, 32, 0, lane);

        acc[0][0] = __builtin_amdgcn_wmma_f32_16x16x32_bf16(false, a0, false, b0,
                                                            (short)0, acc[0][0], false, false);
        acc[0][1] = __builtin_amdgcn_wmma_f32_16x16x32_bf16(false, a0, false, b1,
                                                            (short)0, acc[0][1], false, false);
        acc[1][0] = __builtin_amdgcn_wmma_f32_16x16x32_bf16(false, a1, false, b0,
                                                            (short)0, acc[1][0], false, false);
        acc[1][1] = __builtin_amdgcn_wmma_f32_16x16x32_bf16(false, a1, false, b1,
                                                            (short)0, acc[1][1], false, false);
    }

    // Epilogue: C layout -> element (M = r + 8*half, N = col)
#pragma unroll
    for (int i = 0; i < 2; ++i) {
#pragma unroll
        for (int jj = 0; jj < 2; ++jj) {
#pragma unroll
            for (int r = 0; r < 8; ++r) {
                int e = e0 + wr * 32 + i * 16 + r + 8 * half;
                int n = n0 + wc * 32 + jj * 16 + col;
                float val = acc[i][jj][r] + bias[e];
                int hh = e & 15, hd = e >> 4;
                if (mode == 2) {
                    out[(((size_t)(b * Hc + hh)) * HDc + hd) * Nc + n] = f2bf(val);
                } else {
                    if (mode == 0) val *= 0.125f;  // 1/sqrt(64)
                    out[(((size_t)(b * Hc + hh)) * Nc + n) * HDc + hd] = f2bf(val);
                }
            }
        }
    }
}

// ---------------------------------------------------------------------------
// Kernel 2: flash attention per (b,h). 128 q-rows per WG (8 waves x 16 rows).
// Q,K: [b][h][n][hd] bf16 (Q pre-scaled). V: [b][h][hd][n] bf16.
// K/V tiles staged with GLOBAL_LOAD_ASYNC_TO_LDS_B128 (ASYNCcnt).
// Output X: [b][d][n] bf16 with d = hd*16 + h.
// ---------------------------------------------------------------------------
__global__ __launch_bounds__(256)
void flash_attn_kernel(const unsigned short* __restrict__ Q,
                       const unsigned short* __restrict__ K,
                       const unsigned short* __restrict__ V,
                       unsigned short* __restrict__ Xout) {
    __shared__ __align__(16) unsigned short ldsK[64 * 64];   // [m][hd]
    __shared__ __align__(16) unsigned short ldsV[64 * 64];   // [hd][m]
    __shared__ __align__(16) unsigned short ldsP[8 * 16 * 64]; // per-wave P scratch

    const int q0 = blockIdx.x * 128;
    const int h  = blockIdx.y;
    const int b  = blockIdx.z;
    const int bh = b * Hc + h;
    const int t = threadIdx.x, lane = t & 31, w = t >> 5;
    const int half = lane >> 4, col = lane & 15;
    const int qrow = q0 + w * 16;

    // Q A-fragments for this wave's 16 rows (kept in registers for all key blocks)
    const unsigned short* qbase = Q + ((size_t)bh * Nc + qrow) * HDc;
    v16bf qa0 = load_frag_a(qbase, HDc, 0, lane);
    v16bf qa1 = load_frag_a(qbase, HDc, 32, lane);

    float m_st[8], l_st[8];
    v8f o[4] = {};
#pragma unroll
    for (int r = 0; r < 8; ++r) { m_st[r] = -3.0e38f; l_st[r] = 0.0f; }

    unsigned short* pScr = ldsP + w * (16 * 64);

    for (int kb = 0; kb < Nc; kb += 64) {
        __syncthreads();  // previous iteration's readers done before overwrite

        // --- Async stage K block [m][hd]: contiguous 8 KB ---
        {
            const char* gk = (const char*)(K + ((size_t)bh * Nc + kb) * HDc);
#pragma unroll
            for (int i = 0; i < 2; ++i) {
                int seg = t + i * 256;  // 0..511 x 16B = 8 KB
                async_ld_b128((char*)ldsK + seg * 16, gk + seg * 16);
            }
        }
        // --- Async stage V block [hd][m]: 64 rows of 128 B (global stride 4 KB) ---
        {
#pragma unroll
            for (int i = 0; i < 2; ++i) {
                int idx = t + i * 256;  // 0..511
                int hd = idx >> 3, seg = idx & 7;
                const char* gv = (const char*)(V + ((size_t)bh * HDc + hd) * Nc + kb);
                async_ld_b128((char*)ldsV + hd * 128 + seg * 16, gv + seg * 16);
            }
        }
        wait_async0();
        __syncthreads();

        // S = Q * K^T for 64 keys: 4 sub-tiles of 16 keys
        v8f s[4];
#pragma unroll
        for (int c = 0; c < 4; ++c) {
            v16bf kb0 = load_frag_b_t(ldsK + (c * 16) * 64, 64, 0, lane);
            v16bf kb1 = load_frag_b_t(ldsK + (c * 16) * 64, 64, 32, lane);
            v8f z = {};
            z = __builtin_amdgcn_wmma_f32_16x16x32_bf16(false, qa0, false, kb0,
                                                        (short)0, z, false, false);
            z = __builtin_amdgcn_wmma_f32_16x16x32_bf16(false, qa1, false, kb1,
                                                        (short)0, z, false, false);
            s[c] = z;
        }

        // Online softmax: row stats (rows M = r + 8*half, keys across 16 lanes)
        float newm[8];
#pragma unroll
        for (int r = 0; r < 8; ++r) {
            float mx = fmaxf(fmaxf(s[0][r], s[1][r]), fmaxf(s[2][r], s[3][r]));
#pragma unroll
            for (int off = 1; off < 16; off <<= 1)
                mx = fmaxf(mx, __shfl_xor(mx, off, 32));
            newm[r] = fmaxf(m_st[r], mx);
        }
        float rsum[8];
#pragma unroll
        for (int r = 0; r < 8; ++r) rsum[r] = 0.0f;
#pragma unroll
        for (int c = 0; c < 4; ++c) {
#pragma unroll
            for (int r = 0; r < 8; ++r) {
                float p = __expf(s[c][r] - newm[r]);
                rsum[r] += p;
                pScr[(r + 8 * half) * 64 + c * 16 + col] = f2bf(p);
            }
        }
#pragma unroll
        for (int r = 0; r < 8; ++r) {
#pragma unroll
            for (int off = 1; off < 16; off <<= 1)
                rsum[r] += __shfl_xor(rsum[r], off, 32);
            float alpha = __expf(m_st[r] - newm[r]);
            l_st[r] = l_st[r] * alpha + rsum[r];
            m_st[r] = newm[r];
#pragma unroll
            for (int c = 0; c < 4; ++c) o[c][r] *= alpha;
        }
        __syncthreads();  // P scratch visible (and keep waves in lockstep)

        // O += P(16x64) * V(64keys x 64hd): 2 K-steps x 4 hd tiles
#pragma unroll
        for (int s2 = 0; s2 < 2; ++s2) {
            v16bf pa = load_frag_a(pScr, 64, s2 * 32, lane);
#pragma unroll
            for (int c = 0; c < 4; ++c) {
                v16bf vb = load_frag_b_t(ldsV + (c * 16) * 64, 64, s2 * 32, lane);
                o[c] = __builtin_amdgcn_wmma_f32_16x16x32_bf16(false, pa, false, vb,
                                                               (short)0, o[c], false, false);
            }
        }
    }

    // Normalize and write X[b][d][n], d = hd*16 + h
#pragma unroll
    for (int r = 0; r < 8; ++r) {
        float inv = 1.0f / l_st[r];
        int n = qrow + r + 8 * half;
#pragma unroll
        for (int c = 0; c < 4; ++c) {
            int hd = c * 16 + col;
            int d = hd * Hc + h;
            Xout[((size_t)b * Dc + d) * Nc + n] = f2bf(o[c][r] * inv);
        }
    }
}

// ---------------------------------------------------------------------------
// Kernel 3: output projection. out[b,e,n] = sum_d Wm[e,d]*X[b,d,n] + bm[e] (fp32)
// ---------------------------------------------------------------------------
__global__ __launch_bounds__(256)
void out_proj_kernel(const unsigned short* __restrict__ X, const float* __restrict__ W,
                     const float* __restrict__ bias, float* __restrict__ out) {
    __shared__ __align__(16) unsigned short ldsW[128 * 32];
    __shared__ __align__(16) unsigned short ldsX[64 * 32];

    const int b  = blockIdx.z;
    const int e0 = blockIdx.y * 128;
    const int n0 = blockIdx.x * 64;
    const int t = threadIdx.x, lane = t & 31, w = t >> 5;
    const int wr = w >> 1, wc = w & 1;
    const int half = lane >> 4, col = lane & 15;

    v8f acc[2][2] = {};

    for (int k0 = 0; k0 < Dc; k0 += 32) {
        __syncthreads();
        if (k0 + 32 < Dc) {
            __builtin_prefetch(&W[(size_t)(e0 + (t >> 1)) * Dc + k0 + 32], 0, 0);
            __builtin_prefetch(&X[(size_t)b * Dc * Nc + (size_t)(k0 + 32 + (t >> 3)) * Nc + n0],
                               0, 0);
        }
#pragma unroll
        for (int i = 0; i < 16; ++i) {
            int idx = t + i * 256;
            int r = idx >> 5, c = idx & 31;
            ldsW[idx] = f2bf(W[(size_t)(e0 + r) * Dc + k0 + c]);
        }
#pragma unroll
        for (int i = 0; i < 8; ++i) {
            int idx = t + i * 256;
            int k = idx >> 6, n = idx & 63;
            ldsX[n * 32 + k] = X[(size_t)b * Dc * Nc + (size_t)(k0 + k) * Nc + n0 + n];
        }
        __syncthreads();

        v16bf a0 = load_frag_a(ldsW + (wr * 32) * 32, 32, 0, lane);
        v16bf a1 = load_frag_a(ldsW + (wr * 32 + 16) * 32, 32, 0, lane);
        v16bf b0 = load_frag_b_t(ldsX + (wc * 32) * 32, 32, 0, lane);
        v16bf b1 = load_frag_b_t(ldsX + (wc * 32 + 16) * 32, 32, 0, lane);

        acc[0][0] = __builtin_amdgcn_wmma_f32_16x16x32_bf16(false, a0, false, b0,
                                                            (short)0, acc[0][0], false, false);
        acc[0][1] = __builtin_amdgcn_wmma_f32_16x16x32_bf16(false, a0, false, b1,
                                                            (short)0, acc[0][1], false, false);
        acc[1][0] = __builtin_amdgcn_wmma_f32_16x16x32_bf16(false, a1, false, b0,
                                                            (short)0, acc[1][0], false, false);
        acc[1][1] = __builtin_amdgcn_wmma_f32_16x16x32_bf16(false, a1, false, b1,
                                                            (short)0, acc[1][1], false, false);
    }

#pragma unroll
    for (int i = 0; i < 2; ++i) {
#pragma unroll
        for (int jj = 0; jj < 2; ++jj) {
#pragma unroll
            for (int r = 0; r < 8; ++r) {
                int e = e0 + wr * 32 + i * 16 + r + 8 * half;
                int n = n0 + wc * 32 + jj * 16 + col;
                out[(size_t)b * Dc * Nc + (size_t)e * Nc + n] = acc[i][jj][r] + bias[e];
            }
        }
    }
}

// ---------------------------------------------------------------------------
extern "C" void kernel_launch(void* const* d_in, const int* in_sizes, int n_in,
                              void* d_out, int out_size, void* d_ws, size_t ws_size,
                              hipStream_t stream) {
    const float* query  = (const float*)d_in[0];
    const float* key_in = (const float*)d_in[1];
    const float* value  = (const float*)d_in[2];
    const float* Wq = (const float*)d_in[3];
    const float* bq = (const float*)d_in[4];
    const float* Wk = (const float*)d_in[5];
    const float* bk = (const float*)d_in[6];
    const float* Wv = (const float*)d_in[7];
    const float* bv = (const float*)d_in[8];
    const float* Wm = (const float*)d_in[9];
    const float* bm = (const float*)d_in[10];

    const size_t elems = (size_t)Bc * Dc * Nc;  // 8,388,608 bf16 elems = 16 MB each
    unsigned short* Qws = (unsigned short*)d_ws;
    unsigned short* Kws = Qws + elems;
    unsigned short* Vws = Kws + elems;
    unsigned short* Xws = Vws + elems;

    dim3 gProj(Nc / 64, Dc / 128, Bc);  // (32, 8, 4)
    qkv_proj_kernel<<<gProj, 256, 0, stream>>>(query,  Wq, bq, Qws, 0);
    qkv_proj_kernel<<<gProj, 256, 0, stream>>>(key_in, Wk, bk, Kws, 1);
    qkv_proj_kernel<<<gProj, 256, 0, stream>>>(value,  Wv, bv, Vws, 2);

    dim3 gAttn(Nc / 128, Hc, Bc);       // (16, 16, 4)
    flash_attn_kernel<<<gAttn, 256, 0, stream>>>(Qws, Kws, Vws, Xws);

    out_proj_kernel<<<gProj, 256, 0, stream>>>(Xws, Wm, bm, (float*)d_out);
}
